// KPCN_68607807586710
// MI455X (gfx1250) — matrix operand order repaired and checked by reading, not compiled, not run on hardware
//
#include <hip/hip_runtime.h>
#include <stdint.h>

// ---------------------------------------------------------------------------
// KPCN on MI455X (gfx1250):
//  * 9x VALID 5x5 convs as tap-decomposed implicit GEMM on v_wmma_f32_16x16x32_bf16
//  * activations bf16 NHWC, channels padded to mult-of-32 (K) / mult-of-16 (N)
//  * per-block (MR+4) x 20 x Cin input patch staged via
//    global_load_async_to_lds_b128 (ASYNCcnt + barrier); LDS pixel stride is
//    padded (+8 elems = 4 banks) so A-fragment ds_load_b128 are conflict-free
//  * each wave holds MR A-fragments in registers and reuses every B fragment
//    MR times -> B/L2 traffic cut by MR vs. naive (B is the dominant stream)
//  * weights device-prepacked into WMMA B-fragment-major layout (contig 32B/lane)
//  * f32 accumulate -> bias -> ReLU fused epilogue
//  * final: per-pixel 441-way softmax + reflect-padded 21x21 gather (1 wave/pixel)
// ---------------------------------------------------------------------------

typedef __attribute__((ext_vector_type(16))) __bf16 v16bf;
typedef __attribute__((ext_vector_type(8)))  float  v8f;
typedef unsigned short u16;

__device__ __forceinline__ u16 f32_to_bf16(float f) {
  union { float f; uint32_t u; } v; v.f = f;
  uint32_t u = v.u;
  return (u16)((u + 0x7FFFu + ((u >> 16) & 1u)) >> 16);   // round-to-nearest-even
}
__device__ __forceinline__ float bf16_to_f32(u16 h) {
  union { uint32_t u; float f; } v; v.u = ((uint32_t)h) << 16;
  return v.f;
}

union FragBF { v16bf v; uint4 q[2]; };

// ---- pack fp32 NCHW input -> bf16 NHWC, C padded 28->32 ----------------------
__global__ void k_pack_input(const float* __restrict__ x, u16* __restrict__ act) {
  int idx = blockIdx.x * blockDim.x + threadIdx.x;   // 16*128*128*32
  int c = idx & 31;
  int w = (idx >> 5) & 127;
  int h = (idx >> 12) & 127;
  int n = idx >> 19;
  float v = 0.f;
  if (c < 28) v = x[(((size_t)n * 28 + c) * 128 + h) * 128 + w];
  act[idx] = f32_to_bf16(v);
}

// ---- pack fp32 OIHW weights -> bf16 WMMA B-fragment-major --------------------
// block layout: [(t*KC + q)*NT + j][lane 0..31][slot 0..15], where
//   lane<16 : N = j*16+lane,    K = q*32 + (slot&7) + 16*(slot>>3)
//   lane>=16: N = j*16+lane-16, K = q*32 + 8 + (slot&7) + 16*(slot>>3)
__global__ void k_pack_w(const float* __restrict__ w, u16* __restrict__ dst,
                         int Cin, int Cout, int KC, int NT, int total) {
  int idx = blockIdx.x * blockDim.x + threadIdx.x;
  if (idx >= total) return;
  int slot = idx & 15;
  int lane = (idx >> 4) & 31;
  int rest = idx >> 9;
  int j = rest % NT;
  int tq = rest / NT;
  int q = tq % KC;
  int t = tq / KC;
  int n = j * 16 + (lane & 15);
  int k = q * 32 + ((lane >> 4) << 3) + (slot & 7) + ((slot >> 3) << 4);
  float v = 0.f;
  if (n < Cout && k < Cin) v = w[(((size_t)n * Cin + k) * 25) + t];
  dst[idx] = f32_to_bf16(v);
}

// ---- implicit-GEMM conv ------------------------------------------------------
// block = 128 (4 waves); block computes MR output rows x 16 cols x CoutP chans.
// grid = (ceil(Wout/16), Hout/MR, 16 batch)
template <int MR, int JPW, int KC>
__global__ __launch_bounds__(128)
void k_conv_wmma(const u16* __restrict__ inAct, u16* __restrict__ outAct,
                 const u16* __restrict__ wfrag, const float* __restrict__ bias,
                 int CoutReal, int Hin, int Win, int relu) {
  constexpr int CinP = KC * 32;
  constexpr int PIX  = CinP + 8;          // padded LDS pixel stride (bank spread)
  constexpr int NT   = 4 * JPW;           // total N-tiles (4 waves/block)
  constexpr int CoutP = NT * 16;
  constexpr int ROWS = MR + 4;            // input rows needed
  const int Hout = Hin - 4, Wout = Win - 4;
  const int lane = threadIdx.x & 31;
  const int wave = threadIdx.x >> 5;
  const int w0  = blockIdx.x * 16;
  const int ho0 = blockIdx.y * MR;
  const int nImg = blockIdx.z;

  // ---- stage ROWS x 20 x CinP channels into LDS (async copy, padded) --------
  __shared__ __align__(16) u16 tile[ROWS * 20 * PIX];
  {
    constexpr int VPP = CinP / 8;                 // 16B vectors per pixel
    constexpr int NV  = ROWS * 20 * VPP;          // multiple of 128 for all cfgs
    const uint32_t ldsBase = (uint32_t)(size_t)&tile[0];
    const size_t gRow = ((size_t)nImg * Hin + ho0) * Win;
    for (int v = threadIdx.x; v < NV; v += 128) {
      int pix = v / VPP;
      int k   = v - pix * VPP;
      int row = pix / 20, col = pix - row * 20;
      int gcol = w0 + col; if (gcol > Win - 1) gcol = Win - 1;   // edge-tile clamp
      uint32_t goff = (uint32_t)(((gRow + (size_t)row * Win + gcol) * CinP + k * 8) * 2);
      uint32_t ldsoff = ldsBase + (uint32_t)(pix * PIX + k * 8) * 2;
      // per-lane: LDS[ldsoff] <= MEM[saddr + goff], 16 bytes, tracked by ASYNCcnt
      asm volatile("global_load_async_to_lds_b128 %0, %1, %2"
                   :: "v"(ldsoff), "v"(goff), "s"(inAct) : "memory");
    }
    asm volatile("s_wait_asynccnt 0" ::: "memory");
    __syncthreads();
  }

  const int mrow = lane & 15;
  const int hi   = lane >> 4;
  int mrowc = mrow;                                  // clamp edge-tile output pixels
  if (w0 + mrowc > Wout - 1) mrowc = Wout - 1 - w0;

  v8f acc[MR][JPW];
#pragma unroll
  for (int r = 0; r < MR; ++r)
#pragma unroll
    for (int jj = 0; jj < JPW; ++jj)
      acc[r][jj] = v8f{0.f,0.f,0.f,0.f,0.f,0.f,0.f,0.f};

  for (int t = 0; t < 25; ++t) {
    const int dy = t / 5, dx = t % 5;
#pragma unroll
    for (int q = 0; q < KC; ++q) {
      // MR A fragments: 16-bit 16x32 lane layout -> two 16B LDS loads each
      FragBF a[MR];
#pragma unroll
      for (int r = 0; r < MR; ++r) {
        const u16* ap = &tile[((r + dy) * 20 + mrowc + dx) * PIX + q * 32 + hi * 8];
        a[r].q[0] = *(const uint4*)(ap);
        a[r].q[1] = *(const uint4*)(ap + 16);
      }
      const u16* bb = wfrag + ((size_t)(t * KC + q) * NT + wave * JPW) * 512 + lane * 16;
      __builtin_prefetch(bb + (size_t)NT * 512, 0, 0);   // warm next weight block (L2)
#pragma unroll
      for (int jj = 0; jj < JPW; ++jj) {
        FragBF b;
        const uint4* bp = (const uint4*)(bb + (size_t)jj * 512);
        b.q[0] = bp[0];
        b.q[1] = bp[1];
#pragma unroll
        for (int r = 0; r < MR; ++r)                     // reuse B across MR rows
          acc[r][jj] = __builtin_amdgcn_wmma_f32_16x16x32_bf16(
              false, a[r].v, false, b.v, (short)0, acc[r][jj], false, false);
      }
    }
  }

  // epilogue: bias + ReLU, bf16 NHWC store.  C/D layout: VGPR i -> M = i + 8*hi, N = lane&15
#pragma unroll
  for (int jj = 0; jj < JPW; ++jj) {
    const int nch = (wave * JPW + jj) * 16 + (lane & 15);
    const float bv = (nch < CoutReal) ? bias[nch] : 0.f;
#pragma unroll
    for (int r = 0; r < MR; ++r) {
      const size_t rowBase = ((size_t)nImg * Hout + ho0 + r) * Wout;
#pragma unroll
      for (int i = 0; i < 8; ++i) {
        const int wo = w0 + i + hi * 8;
        if (wo < Wout) {
          float v = acc[r][jj][i] + bv;
          if (relu && v < 0.f) v = 0.f;
          outAct[(rowBase + wo) * CoutP + nch] = f32_to_bf16(v);
        }
      }
    }
  }
}

// ---- softmax over 441 logits + reflect-padded 21x21 weighted gather ----------
__global__ __launch_bounds__(256)
void k_apply(const u16* __restrict__ logits, const float* __restrict__ inp,
             float* __restrict__ out) {
  const int lane = threadIdx.x & 31;
  const int wave = threadIdx.x >> 5;
  const int pix = blockIdx.x * 8 + wave;
  if (pix >= 16 * 92 * 92) return;
  const int wo = pix % 92;
  const int ho = (pix / 92) % 92;
  const int n  = pix / (92 * 92);

  const u16* lg = logits + (size_t)pix * 448;   // padded C=448, use first 441
  float l[14];
  float mx = -1e30f;
#pragma unroll
  for (int i = 0; i < 14; ++i) {
    int ch = lane + i * 32;
    float v = (ch < 441) ? bf16_to_f32(lg[ch]) : -1e30f;
    l[i] = v;
    mx = fmaxf(mx, v);
  }
#pragma unroll
  for (int off = 16; off > 0; off >>= 1) mx = fmaxf(mx, __shfl_xor(mx, off, 32));

  const float* inpn = inp + (size_t)n * 3 * 128 * 128;
  float s = 0.f, a0 = 0.f, a1 = 0.f, a2 = 0.f;
#pragma unroll
  for (int i = 0; i < 14; ++i) {
    int ch = lane + i * 32;
    if (ch < 441) {
      float wgt = __expf(l[i] - mx);
      s += wgt;
      int dy = ch / 21, dx = ch - dy * 21;
      int py = ho + dy - 10; py = py < 0 ? -py : (py > 91 ? 182 - py : py);
      int px = wo + dx - 10; px = px < 0 ? -px : (px > 91 ? 182 - px : px);
      size_t base = (size_t)(18 + py) * 128 + (18 + px);   // crop offset 18
      a0 += wgt * inpn[base];
      a1 += wgt * inpn[(size_t)128 * 128 + base];
      a2 += wgt * inpn[(size_t)2 * 128 * 128 + base];
    }
  }
#pragma unroll
  for (int off = 16; off > 0; off >>= 1) {
    s  += __shfl_xor(s,  off, 32);
    a0 += __shfl_xor(a0, off, 32);
    a1 += __shfl_xor(a1, off, 32);
    a2 += __shfl_xor(a2, off, 32);
  }
  if (lane == 0) {
    float inv = 1.f / s;
    size_t o = ((size_t)n * 3) * 92 * 92 + (size_t)ho * 92 + wo;
    out[o]                 = a0 * inv;
    out[o + 92 * 92]       = a1 * inv;
    out[o + 2 * 92 * 92]   = a2 * inv;
  }
}

extern "C" void kernel_launch(void* const* d_in, const int* in_sizes, int n_in,
                              void* d_out, int out_size, void* d_ws, size_t ws_size,
                              hipStream_t stream) {
  const float* x   = (const float*)d_in[0];
  const float* inp = (const float*)d_in[1];
  // d_in[2+2l] = w_l (OIHW fp32), d_in[3+2l] = b_l (fp32)

  char* ws = (char*)d_ws;
  u16* wbuf = (u16*)ws;                                   // ~8.8 MB packed weights
  u16* buf0 = (u16*)(ws + (size_t)16 * 1024 * 1024);      // 64 MB ping
  u16* buf1 = (u16*)(ws + (size_t)80 * 1024 * 1024);      // 128 MB pong (holds conv9: 121 MB)

  struct L { int Cin, Cout, KC, NT, MR, Hin, relu; };
  L ls[9];
  ls[0] = {28, 100, 1, 8, 4, 128, 1};
  for (int l = 1; l < 8; ++l) ls[l] = {100, 100, 4, 8, 4, 128 - 4 * l, 1};
  ls[8] = {100, 441, 4, 28, 2, 96, 0};

  k_pack_input<<<(16 * 128 * 128 * 32) / 256, 256, 0, stream>>>(x, buf0);

  size_t woff = 0;
  u16* cur = buf0; u16* nxt = buf1;
  for (int l = 0; l < 9; ++l) {
    const L& c = ls[l];
    const float* wl = (const float*)d_in[2 + 2 * l];
    const float* bl = (const float*)d_in[3 + 2 * l];
    const int total = 25 * c.KC * c.NT * 512;
    u16* wdst = wbuf + woff;
    k_pack_w<<<(total + 255) / 256, 256, 0, stream>>>(wl, wdst, c.Cin, c.Cout, c.KC, c.NT, total);

    const int Hout = c.Hin - 4, Wout = Hout;
    dim3 grid((Wout + 15) / 16, (unsigned)(Hout / c.MR), 16);
    if (l == 0)
      k_conv_wmma<4, 2, 1><<<grid, 128, 0, stream>>>(cur, nxt, wdst, bl, c.Cout, c.Hin, c.Hin, c.relu);
    else if (l < 8)
      k_conv_wmma<4, 2, 4><<<grid, 128, 0, stream>>>(cur, nxt, wdst, bl, c.Cout, c.Hin, c.Hin, c.relu);
    else
      k_conv_wmma<2, 7, 4><<<grid, 128, 0, stream>>>(cur, nxt, wdst, bl, c.Cout, c.Hin, c.Hin, c.relu);

    woff += (size_t)total;
    u16* t = cur; cur = nxt; nxt = t;
  }

  // cur == buf1 holds conv9 logits (16,92,92,448) bf16
  k_apply<<<(16 * 92 * 92) / 8, 256, 0, stream>>>(cur, inp, (float*)d_out);
}
